// TMB_11793980195104
// MI455X (gfx1250) — matrix operand order repaired and checked by reading, not compiled, not run on hardware
//
#include <hip/hip_runtime.h>
#include <hip/hip_bf16.h>
#include <math.h>

// ---------------------------------------------------------------------------
// Problem constants (reference: B=4, C=128, H=W=64 -> N=4096)
// ---------------------------------------------------------------------------
#define BATCH 4
#define CH    128
#define NPIX  4096
#define THITA 1.0e-4f

typedef __attribute__((ext_vector_type(16))) __bf16          v16bf;
typedef __attribute__((ext_vector_type(16))) unsigned short  v16u;
typedef __attribute__((ext_vector_type(8)))  unsigned short  v8u;
typedef __attribute__((ext_vector_type(8)))  float           v8f;

// float -> bf16 bits, round-to-nearest-even
__device__ __forceinline__ unsigned short f2bf(float f) {
  unsigned u = __builtin_bit_cast(unsigned, f);
  unsigned r = u + 0x7FFFu + ((u >> 16) & 1u);
  return (unsigned short)(r >> 16);
}

// ---------------------------------------------------------------------------
// WMMA fragment builders. All gathers are two contiguous 16-byte LDS loads
// per lane (-> ds_load_b128 x2).
// A 16x32 (ISA 7.12.2): lane m=lane&15, half=lane>>4; two contiguous runs
//   k in [half*8, half*8+8) and [16+half*8, ...+8). Needs row-major [m][k].
// B 32x16: lane n=lane&15; same two runs as A. Needs [n][k].
// ---------------------------------------------------------------------------
__device__ __forceinline__ v16bf cat2(v8u lo, v8u hi) {
  v16u t = __builtin_shufflevector(lo, hi, 0, 1, 2, 3, 4, 5, 6, 7,
                                   8, 9, 10, 11, 12, 13, 14, 15);
  return __builtin_bit_cast(v16bf, t);
}
__device__ __forceinline__ v16bf frag_a(const unsigned short* base, int ld,
                                        int m, int half) {
  const unsigned short* p = base + m * ld + half * 8;
  return cat2(*(const v8u*)(p), *(const v8u*)(p + 16));
}
__device__ __forceinline__ v16bf frag_b(const unsigned short* base, int ld,
                                        int n, int half) {
  const unsigned short* p = base + n * ld + half * 8;
  return cat2(*(const v8u*)(p), *(const v8u*)(p + 16));
}

// ---------------------------------------------------------------------------
// Tiny convs (K=4 and K=2): scalar FMA, memory-bound, WMMA pointless here.
// ---------------------------------------------------------------------------
__global__ __launch_bounds__(256) void conv_neighbor(
    const float* __restrict__ ev, const float* __restrict__ w,
    const float* __restrict__ bias, float* __restrict__ out) {
  size_t i = (size_t)blockIdx.x * 256 + threadIdx.x;  // B*256*NPIX
  size_t p = i & (NPIX - 1);
  size_t o = (i >> 12) & 255;
  size_t b = i >> 20;
  const float* e = ev + b * 6 * NPIX;
  float v = bias[o] + w[o * 4 + 0] * e[0 * NPIX + p]
                    + w[o * 4 + 1] * e[1 * NPIX + p]
                    + w[o * 4 + 2] * e[4 * NPIX + p]
                    + w[o * 4 + 3] * e[5 * NPIX + p];
  out[(b * 256 + o) * NPIX + p] = v > 0.f ? v : 0.f;
}

__global__ __launch_bounds__(256) void conv_now1(
    const float* __restrict__ ev, const float* __restrict__ w,
    const float* __restrict__ bias, float* __restrict__ out) {
  size_t i = (size_t)blockIdx.x * 256 + threadIdx.x;  // B*128*NPIX
  size_t p = i & (NPIX - 1);
  size_t o = (i >> 12) & 127;
  size_t b = i >> 19;
  const float* e = ev + b * 6 * NPIX;
  float v = bias[o] + w[o * 2 + 0] * e[2 * NPIX + p]
                    + w[o * 2 + 1] * e[3 * NPIX + p];
  out[(b * 128 + o) * NPIX + p] = v > 0.f ? v : 0.f;
}

// ---------------------------------------------------------------------------
// Conv1x1 as bf16 WMMA GEMM: Y = ReLU(W[128,Cin] * X[Cin,NPIX] + b)
// 8 waves x (16M x 64N) tiles; K step 32; next X tile prefetched while the
// current one is being consumed (global_prefetch_b8).
// Output modes:
//   0: fp32 into [YC][NPIX] at channel offset yoff (chain intermediates)
//   1: bf16 TRANSPOSED [NPIX][128] (q, k) -- one packed b128 store per tile
//   2: bf16 [128][NPIX]            (v)
// grid = (NPIX/64, 1, BATCH), block = 256
// ---------------------------------------------------------------------------
__global__ __launch_bounds__(256) void gemm128(
    const float* __restrict__ X, int XC, int xoff, int Cin,
    const float* __restrict__ W, const float* __restrict__ bias,
    void* __restrict__ Yv, int YC, int yoff, int mode) {
  __shared__ __align__(16) unsigned short Wl[128 * 32];  // [o][k]
  __shared__ __align__(16) unsigned short Xt[64 * 40];   // [n][k], padded
  const int tid = threadIdx.x;
  const int wave = tid >> 5, lane = tid & 31;
  const int half = lane >> 4, l16 = lane & 15;
  const int n0 = blockIdx.x * 64;
  const size_t b = blockIdx.z;
  const float* Xb = X + ((size_t)b * XC + xoff) * NPIX;

  v8f acc[4] = {};
  for (int k0 = 0; k0 < Cin; k0 += 32) {
#pragma unroll
    for (int t = 0; t < 16; ++t) {               // 128x32 weight tile
      int idx = tid + t * 256;
      int o = idx >> 5, kk = idx & 31;
      Wl[idx] = f2bf(W[(size_t)o * Cin + k0 + kk]);
    }
#pragma unroll
    for (int t = 0; t < 8; ++t) {                // 32x64 activation tile
      int idx = tid + t * 256;
      int kk = idx >> 6, n = idx & 63;           // coalesced over n
      Xt[n * 40 + kk] = f2bf(Xb[(size_t)(k0 + kk) * NPIX + n0 + n]);
    }
    if (k0 + 32 < Cin) {                         // prefetch next X tile
      int kk = tid >> 3, nn = (tid & 7) * 8;     // 32 rows x 8 slices
      __builtin_prefetch(&Xb[(size_t)(k0 + 32 + kk) * NPIX + n0 + nn], 0, 3);
    }
    __syncthreads();
    v16bf a = frag_a(Wl, 32, wave * 16 + l16, half);
#pragma unroll
    for (int s = 0; s < 4; ++s) {
      v16bf bb = frag_b(Xt, 40, s * 16 + l16, half);
      acc[s] = __builtin_amdgcn_wmma_f32_16x16x32_bf16(
          false, a, false, bb, (short)0, acc[s], false, false);
    }
    __syncthreads();
  }

  if (mode == 0) {
    float* Yb = (float*)Yv + ((size_t)b * YC + yoff) * NPIX;
#pragma unroll
    for (int s = 0; s < 4; ++s) {
      int n = n0 + s * 16 + l16;
#pragma unroll
      for (int r = 0; r < 8; ++r) {
        int m = wave * 16 + r + 8 * half;        // C/D layout: VGPR r -> M
        float v = acc[s][r] + bias[m];
        Yb[(size_t)m * NPIX + n] = v > 0.f ? v : 0.f;
      }
    }
  } else if (mode == 1) {
    // bf16 transposed [NPIX][128]: rows m = wave*16+8*half+r are contiguous
    unsigned short* Yt = (unsigned short*)Yv + (size_t)b * NPIX * 128;
    const int mb = wave * 16 + 8 * half;
#pragma unroll
    for (int s = 0; s < 4; ++s) {
      int n = n0 + s * 16 + l16;
      v8u u;
#pragma unroll
      for (int r = 0; r < 8; ++r) {
        float v = acc[s][r] + bias[mb + r];
        u[r] = f2bf(v > 0.f ? v : 0.f);
      }
      *(v8u*)(&Yt[(size_t)n * 128 + mb]) = u;    // single 16B store
    }
  } else {
    unsigned short* Yb = (unsigned short*)Yv + (size_t)b * 128 * NPIX;
#pragma unroll
    for (int s = 0; s < 4; ++s) {
      int n = n0 + s * 16 + l16;
#pragma unroll
      for (int r = 0; r < 8; ++r) {
        int m = wave * 16 + r + 8 * half;
        float v = acc[s][r] + bias[m];
        Yb[(size_t)m * NPIX + n] = f2bf(v > 0.f ? v : 0.f);
      }
    }
  }
}

// ---------------------------------------------------------------------------
// Flash attention: S = Q^T K (C=128 contraction), online softmax over m,
// O = V P^T, out = THITA*O/l + now_val.  Never materializes the 4096x4096
// attention matrix in HBM.  q/k arrive bf16 pre-transposed [NPIX][128] and
// v bf16 [128][NPIX], so ALL staging is pure 16-byte copies
// (global_load_b128 -> ds_store_b128); next tiles are prefetched while the
// current tile computes (global_prefetch_b8, tiles live in the 192MB L2).
// grid = (NPIX/64, BATCH), block = 256 (8 waves).
// LDS layouts (every fragment gather = 2x ds_load_b128):
//   Qt[n][c] (A of S), Kt[m][c] (B of S), Vl[c][m] (A of O), Pl[n][m] (B of O)
// ---------------------------------------------------------------------------
__global__ __launch_bounds__(256) void flash_attn(
    const unsigned short* __restrict__ Qg,   // [B][NPIX][128] bf16
    const unsigned short* __restrict__ Kg,   // [B][NPIX][128] bf16
    const unsigned short* __restrict__ Vg,   // [B][128][NPIX] bf16
    const float* __restrict__ NowVal, float* __restrict__ Out) {
  __shared__ __align__(16) unsigned short Qt[64 * 136];  // [n][c] padded
  __shared__ __align__(16) unsigned short Kt[32 * 136];  // [m][c] padded
  __shared__ __align__(16) unsigned short Vl[128 * 32];  // [c][m]
  __shared__ __align__(16) unsigned short Pl[64 * 32];   // [n][m]
  __shared__ float Sl[64 * 33];
  __shared__ float mrow[64], lrow[64], arow[64];

  const int tid = threadIdx.x, wave = tid >> 5, lane = tid & 31;
  const int half = lane >> 4, l16 = lane & 15;
  const int n0 = blockIdx.x * 64;
  const size_t b = blockIdx.y;
  const unsigned short* Qb = Qg + b * (size_t)NPIX * 128;
  const unsigned short* Kb = Kg + b * (size_t)NPIX * 128;
  const unsigned short* Vb = Vg + b * (size_t)CH * NPIX;

#pragma unroll
  for (int t = 0; t < 4; ++t) {                // stage 64x128 Q tile once
    int idx = tid + t * 256;                   // 1024 x 16B chunks
    int n = idx >> 4, cc = (idx & 15) * 8;
    *(v8u*)(&Qt[n * 136 + cc]) =
        *(const v8u*)(&Qb[(size_t)(n0 + n) * 128 + cc]);
  }
  if (tid < 64) { mrow[tid] = -3.0e38f; lrow[tid] = 0.f; }
  v8f oacc[4] = {};
  __syncthreads();

  const int nsub = wave & 3, msub = wave >> 2;
  for (int mt = 0; mt < NPIX / 32; ++mt) {
    const int m0 = mt * 32;
#pragma unroll
    for (int t = 0; t < 2; ++t) {              // stage 32x128 K, 128x32 V
      int idx = tid + t * 256;                 // 512 x 16B chunks each
      int m = idx >> 4, cc = (idx & 15) * 8;
      *(v8u*)(&Kt[m * 136 + cc]) =
          *(const v8u*)(&Kb[(size_t)(m0 + m) * 128 + cc]);
      int c = idx >> 2, mm = (idx & 3) * 8;
      *(v8u*)(&Vl[c * 32 + mm]) =
          *(const v8u*)(&Vb[(size_t)c * NPIX + m0 + mm]);
    }
    if (m0 + 32 < NPIX) {                      // prefetch next K/V tiles
      const int m1 = m0 + 32;
      // K: 32 rows x 256B -> 256 threads x 32B slices
      __builtin_prefetch(&Kb[(size_t)(m1 + (tid >> 3)) * 128 + (tid & 7) * 16],
                         0, 3);
      // V: 128 rows x 64B -> 256 threads x 32B slices
      __builtin_prefetch(&Vb[(size_t)(tid >> 1) * NPIX + m1 + (tid & 1) * 16],
                         0, 3);
    }
    __syncthreads();

    // ---- S tile: one 16x16 per wave, K-loop over channels ----
    v8f s = {};
#pragma unroll
    for (int c0 = 0; c0 < CH; c0 += 32) {
      v16bf a  = frag_a(Qt + c0, 136, nsub * 16 + l16, half);  // Q^T rows
      v16bf bb = frag_b(Kt + c0, 136, msub * 16 + l16, half);  // K cols
      s = __builtin_amdgcn_wmma_f32_16x16x32_bf16(
          false, a, false, bb, (short)0, s, false, false);
    }
#pragma unroll
    for (int r = 0; r < 8; ++r)
      Sl[(nsub * 16 + r + 8 * half) * 33 + msub * 16 + l16] = s[r];
    __syncthreads();

    // ---- online softmax: one thread per q-row (waves 0-1 active) ----
    if (tid < 64) {
      float mold = mrow[tid], mnew = mold;
#pragma unroll
      for (int j = 0; j < 32; ++j) {
        float x = Sl[tid * 33 + j];
        mnew = x > mnew ? x : mnew;
      }
      float alpha = __expf(mold - mnew);
      float sum = 0.f;
#pragma unroll
      for (int j = 0; j < 32; ++j) {
        float p = __expf(Sl[tid * 33 + j] - mnew);
        Pl[tid * 32 + j] = f2bf(p);
        sum += p;
      }
      lrow[tid] = lrow[tid] * alpha + sum;
      mrow[tid] = mnew;
      arow[tid] = alpha;
    }
    __syncthreads();

    // ---- rescale O and accumulate O += V * P^T ----
    v16bf av = frag_a(Vl, 32, wave * 16 + l16, half);   // V[c][m]
#pragma unroll
    for (int ss = 0; ss < 4; ++ss) {
      float al = arow[ss * 16 + l16];
#pragma unroll
      for (int r = 0; r < 8; ++r) oacc[ss][r] *= al;
      v16bf bb = frag_b(Pl, 32, ss * 16 + l16, half);   // P^T cols
      oacc[ss] = __builtin_amdgcn_wmma_f32_16x16x32_bf16(
          false, av, false, bb, (short)0, oacc[ss], false, false);
    }
    __syncthreads();   // protect Kt/Vl/Pl before next stage
  }

  const float* NVb = NowVal + b * (size_t)CH * NPIX;
  float* Ob = Out + b * (size_t)CH * NPIX;
#pragma unroll
  for (int ss = 0; ss < 4; ++ss) {
    int n = n0 + ss * 16 + l16;
    float linv = 1.0f / lrow[ss * 16 + l16];
#pragma unroll
    for (int r = 0; r < 8; ++r) {
      int c = wave * 16 + r + 8 * half;
      Ob[(size_t)c * NPIX + n] =
          THITA * (oacc[ss][r] * linv) + NVb[(size_t)c * NPIX + n];
    }
  }
}

// ---------------------------------------------------------------------------
// Host-side orchestration
// ---------------------------------------------------------------------------
extern "C" void kernel_launch(void* const* d_in, const int* in_sizes, int n_in,
                              void* d_out, int out_size, void* d_ws,
                              size_t ws_size, hipStream_t stream) {
  (void)in_sizes; (void)n_in; (void)out_size; (void)ws_size;
  const float* ev        = (const float*)d_in[0];
  const float* w_neigh   = (const float*)d_in[1];
  const float* b_neigh   = (const float*)d_in[2];
  const float* w_prokey  = (const float*)d_in[3];
  const float* b_prokey  = (const float*)d_in[4];
  const float* w_proval  = (const float*)d_in[5];
  const float* b_proval  = (const float*)d_in[6];
  const float* w_latkey  = (const float*)d_in[7];
  const float* b_latkey  = (const float*)d_in[8];
  const float* w_latval  = (const float*)d_in[9];
  const float* b_latval  = (const float*)d_in[10];
  const float* w_tmp1    = (const float*)d_in[11];
  const float* b_tmp1    = (const float*)d_in[12];
  const float* w_now1    = (const float*)d_in[13];
  const float* b_now1    = (const float*)d_in[14];
  const float* w_nowkey  = (const float*)d_in[15];
  const float* b_nowkey  = (const float*)d_in[16];
  const float* w_nowval  = (const float*)d_in[17];
  const float* b_nowval  = (const float*)d_in[18];
  const float* w_q       = (const float*)d_in[19];
  const float* b_q       = (const float*)d_in[20];
  const float* w_k       = (const float*)d_in[21];
  const float* b_k       = (const float*)d_in[22];
  const float* w_v       = (const float*)d_in[23];
  const float* b_v       = (const float*)d_in[24];

  float* ws = (float*)d_ws;
  const size_t PER = (size_t)BATCH * CH * NPIX;  // 2,097,152 floats
  float* neigh  = ws;                 // [B,256,N]  (2*PER)
  float* keycat = ws + 2 * PER;       // [B,256,N]
  float* valcat = ws + 4 * PER;       // [B,256,N]
  float* keyn   = ws + 6 * PER;       // [B,128,N]
  float* valn   = ws + 7 * PER;       // [B,128,N]
  float* nowf   = ws + 8 * PER;       // [B,128,N]
  // reuse after producers are done reading:
  float* nowkey = keycat;             // free after keyn
  float* nowval = keycat + PER;
  unsigned short* q_t = (unsigned short*)neigh;          // bf16 [B][N][128]
  unsigned short* k_t = (unsigned short*)neigh + PER;    // bf16 [B][N][128]
  unsigned short* v_b = (unsigned short*)valcat;         // bf16 [B][128][N]

  conv_neighbor<<<(BATCH * 256 * NPIX) / 256, 256, 0, stream>>>(
      ev, w_neigh, b_neigh, neigh);
  conv_now1<<<(BATCH * 128 * NPIX) / 256, 256, 0, stream>>>(
      ev, w_now1, b_now1, nowf);

  dim3 g(NPIX / 64, 1, BATCH);
  gemm128<<<g, 256, 0, stream>>>(neigh, 256, 0,   128, w_prokey, b_prokey, keycat, 256, 0,   0);
  gemm128<<<g, 256, 0, stream>>>(neigh, 256, 0,   128, w_latkey, b_latkey, keycat, 256, 128, 0);
  gemm128<<<g, 256, 0, stream>>>(neigh, 256, 128, 128, w_proval, b_proval, valcat, 256, 0,   0);
  gemm128<<<g, 256, 0, stream>>>(neigh, 256, 128, 128, w_latval, b_latval, valcat, 256, 128, 0);
  gemm128<<<g, 256, 0, stream>>>(keycat, 256, 0,  256, w_tmp1,   b_tmp1,   keyn,   128, 0,   0);
  gemm128<<<g, 256, 0, stream>>>(valcat, 256, 0,  256, w_tmp1,   b_tmp1,   valn,   128, 0,   0);
  gemm128<<<g, 256, 0, stream>>>(nowf,   128, 0,  128, w_nowkey, b_nowkey, nowkey, 128, 0,   0);
  gemm128<<<g, 256, 0, stream>>>(nowf,   128, 0,  128, w_nowval, b_nowval, nowval, 128, 0,   0);
  gemm128<<<g, 256, 0, stream>>>(keyn,   128, 0,  128, w_q,      b_q,      q_t,    0,   0,   1);
  gemm128<<<g, 256, 0, stream>>>(nowkey, 128, 0,  128, w_k,      b_k,      k_t,    0,   0,   1);
  gemm128<<<g, 256, 0, stream>>>(valn,   128, 0,  128, w_v,      b_v,      v_b,    0,   0,   2);

  flash_attn<<<dim3(NPIX / 64, BATCH), 256, 0, stream>>>(
      q_t, k_t, v_b, nowval, (float*)d_out);
}
